// RayObsGraph_22548578304422
// MI455X (gfx1250) — matrix operand order repaired and checked by reading, not compiled, not run on hardware
//
#include <hip/hip_runtime.h>
#include <hip/hip_bf16.h>
#include <math.h>

// Problem constants (from reference): B=512, T=64, N=128, D=256, H=64, NUM_OUT=18
#define Bn 512
#define Tt 64
#define Nn 128
#define Dd 256
#define Hh 64
#define NO 18

typedef __attribute__((ext_vector_type(16))) __bf16 v16bf;
typedef __attribute__((ext_vector_type(8)))  float  v8f;

union Frag { v16bf v; uint4 q[2]; };

// A-matrix 16x32 bf16 fragment from row-major LDS (stride in elements).
// Lanes 0-15: M=lane, K 0..7 (VGPR0-3) and 16..23 (VGPR4-7);
// lanes 16-31: M=lane-16, K 8..15 and 24..31.
__device__ __forceinline__ v16bf load_tile_a(const __bf16* src, int stride, int lane) {
  const int m  = lane & 15;
  const int kb = (lane < 16) ? 0 : 8;
  Frag f;
  f.q[0] = *(const uint4*)(src + m * stride + kb);        // K = kb .. kb+7
  f.q[1] = *(const uint4*)(src + m * stride + kb + 16);   // K = kb+16 .. kb+23
  return f.v;
}

// B-matrix 32x16 bf16 fragment from column-major LDS storage [n][k]
// (stride = elements per n-row). Lanes 0-15: N=lane, K 0..7 and 16..23; lanes 16-31: K 8..15 and 24..31.
__device__ __forceinline__ v16bf load_tile_b(const __bf16* src, int stride, int lane) {
  const int n  = lane & 15;
  const int kb = (lane < 16) ? 0 : 8;
  Frag f;
  f.q[0] = *(const uint4*)(src + n * stride + kb);
  f.q[1] = *(const uint4*)(src + n * stride + kb + 16);
  return f.v;
}

// C/D 16x16 f32 accumulator -> COLUMN-major bf16 LDS [n][m] (stride Nn).
// Lane holds column n = lane&15, rows mo..mo+7 -> contiguous in m: one b128 store.
__device__ __forceinline__ void store_tile_cm_bf16(__bf16* dst, int lane,
                                                   v8f c, bool do_tanh) {
  const int n  = lane & 15;
  const int mo = (lane < 16) ? 0 : 8;
  union { __bf16 h[8]; uint4 q; } p;
#pragma unroll
  for (int r = 0; r < 8; ++r) {
    float x = c[r];
    if (do_tanh) x = tanhf(x);
    p.h[r] = (__bf16)x;
  }
  *(uint4*)(dst + n * Nn + mo) = p.q;
}

// C/D 16x16 f32 accumulator -> row-major bf16 LDS tile (scalar stores; init only).
__device__ __forceinline__ void store_tile_rm_bf16(__bf16* dst, int stride, int lane,
                                                   v8f c) {
  const int n  = lane & 15;
  const int mo = (lane < 16) ? 0 : 8;
#pragma unroll
  for (int r = 0; r < 8; ++r)
    dst[(mo + r) * stride + n] = (__bf16)c[r];
}

__launch_bounds__(256)
__global__ void rayobs_graph_scan_kernel(
    const float* __restrict__ flat,      // [B*T, D]
    const float* __restrict__ nodes,     // [B, N, D]
    const float* __restrict__ adj,       // [B, N, N]
    const int*   __restrict__ num_nodes, // [B]
    const float* __restrict__ fc_w,      // [D, H]
    const float* __restrict__ fc_b,      // [H]
    const float* __restrict__ w1_rel,    // [H, H]
    const float* __restrict__ w1_root,   // [H, H]
    const float* __restrict__ b1,        // [H]
    const float* __restrict__ w2_rel,    // [H, H]
    const float* __restrict__ w2_root,   // [H, H]
    const float* __restrict__ b2,        // [H]
    const float* __restrict__ logit_w,   // [H, NO]
    const float* __restrict__ logit_b,   // [NO]
    const float* __restrict__ val_w,     // [H, 1]
    const float* __restrict__ val_b,     // [1]
    float* __restrict__ out)             // [B*T, NO+1]
{
  const int b    = blockIdx.x;
  const int tid  = threadIdx.x;
  const int lane = tid & 31;
  const int wave = tid >> 5;       // 8 waves; wave w owns rows [16w, 16w+16)
  const int m0   = wave * 16;

  // LDS (~144.5 KB of the 320 KB WGP pool)
  __shared__ alignas(16) __bf16 s_adj [Nn * Nn];      // adj, row-major (A-tiles + row nn)
  __shared__ alignas(16) __bf16 s_hrm [Nn * Hh];      // h row-major [m][n]        (A-tiles)
  __shared__ alignas(16) __bf16 s_hw  [Hh * Nn];      // h@w1_rel, col-major [n][m] (B-tiles)
  __shared__ alignas(16) __bf16 s_h1  [Hh * Nn];      // tanh(layer1), col-major [n][m]
  __shared__ alignas(16) __bf16 s_w1rel_t [Hh * Hh];  // w1_rel^T  [n][k] (B-tiles)
  __shared__ alignas(16) __bf16 s_w1root_t[Hh * Hh];  // w1_root^T [n][k] (B-tiles)
  __shared__ alignas(16) __bf16 s_fcw_t[Hh * Dd];     // fc_w^T [n][d] (B-tiles + row update)
  __shared__ alignas(16) __bf16 s_w2rel [Hh * Hh];    // w2_rel  [k][j]
  __shared__ alignas(16) __bf16 s_w2root[Hh * Hh];    // w2_root [k][j]
  __shared__ float s_vec [Hh];                        // agg2 row
  __shared__ float s_orow[Hh];                        // layer2 output row

  // ---- one-time setup: stage adj + all weights into LDS (bf16) ----
  const float* adj_b = adj + (size_t)b * Nn * Nn;
  for (int i = tid; i < Nn * Nn; i += 256) s_adj[i] = (__bf16)adj_b[i];

  for (int i = tid; i < Hh * Hh; i += 256) {
    const int k = i >> 6, n = i & 63;
    s_w1rel_t [n * Hh + k] = (__bf16)w1_rel [i];
    s_w1root_t[n * Hh + k] = (__bf16)w1_root[i];
    s_w2rel [i] = (__bf16)w2_rel [i];
    s_w2root[i] = (__bf16)w2_root[i];
  }
  for (int i = tid; i < Dd * Hh; i += 256) {
    const int k = i >> 6, n = i & 63;          // fc_w[k][n], k < 256
    s_fcw_t[n * Dd + k] = (__bf16)fc_w[i];
  }
  const int nn0 = num_nodes[b];
  __syncthreads();

  // Hoist loop-invariant layer-1 weight B-fragments into registers.
  v16bf wrel[2][4], wroot[2][4];
#pragma unroll
  for (int kc = 0; kc < 2; ++kc)
#pragma unroll
    for (int i = 0; i < 4; ++i) {
      wrel [kc][i] = load_tile_b(s_w1rel_t  + (i * 16) * Hh + kc * 32, Hh, lane);
      wroot[kc][i] = load_tile_b(s_w1root_t + (i * 16) * Hh + kc * 32, Hh, lane);
    }
  float bias1[4];
#pragma unroll
  for (int i = 0; i < 4; ++i) bias1[i] = b1[i * 16 + (lane & 15)];

  // ---- initial h = nodes[b] @ fc_w + fc_b via WMMA ----
  // Stream nodes in K-chunks of 32 through a bf16 staging buffer (reuse s_hw,
  // which is dead until stage A of the scan loop).
  {
    const float* nd = nodes + (size_t)b * Nn * Dd;
    __bf16* s_stage = s_hw;                    // [128][32] bf16 = 8 KB
    v8f acc[4];
#pragma unroll
    for (int i = 0; i < 4; ++i) {
      const float bv = fc_b[i * 16 + (lane & 15)];
#pragma unroll
      for (int r = 0; r < 8; ++r) acc[i][r] = bv;
    }
    for (int kc = 0; kc < Dd / 32; ++kc) {     // 8 chunks
      for (int i = tid; i < Nn * 32; i += 256) {
        const int m = i >> 5, d = i & 31;
        s_stage[m * 32 + d] = (__bf16)nd[m * Dd + kc * 32 + d];
      }
      __syncthreads();
      const v16bf a = load_tile_a(s_stage + m0 * 32, 32, lane);
#pragma unroll
      for (int i = 0; i < 4; ++i) {
        const v16bf bt = load_tile_b(s_fcw_t + (i * 16) * Dd + kc * 32, Dd, lane);
        acc[i] = __builtin_amdgcn_wmma_f32_16x16x32_bf16(
            false, a, false, bt, (short)0, acc[i], false, false);
      }
      __syncthreads();                         // stage buffer reuse
    }
#pragma unroll
    for (int i = 0; i < 4; ++i)
      store_tile_rm_bf16(s_hrm + m0 * Hh + i * 16, Hh, lane, acc[i]);
  }
  __syncthreads();

  // Peeled row update for t = 0: h[nn0] = flat[b*T] @ fc_w + fc_b
  if (tid < Hh) {
    const int j = tid;
    const float* xr = flat + (size_t)b * Tt * Dd;
    float acc = fc_b[j];
    for (int d = 0; d < Dd; ++d) acc += xr[d] * (float)s_fcw_t[j * Dd + d];
    s_hrm[((nn0 & (Nn - 1)) * Hh) + j] = (__bf16)acc;
  }
  __syncthreads();

  for (int t = 0; t < Tt; ++t) {
    const int nn = (nn0 + t) & (Nn - 1);

    // (A) hw = h @ w1_rel   [128,64]x[64,64]; reassociated form of (adj@h)@w1_rel
    {
      v8f acc[4];
#pragma unroll
      for (int i = 0; i < 4; ++i)
#pragma unroll
        for (int r = 0; r < 8; ++r) acc[i][r] = 0.0f;

#pragma unroll
      for (int kc = 0; kc < 2; ++kc) {
        const v16bf a = load_tile_a(s_hrm + m0 * Hh + kc * 32, Hh, lane);
#pragma unroll
        for (int i = 0; i < 4; ++i)
          acc[i] = __builtin_amdgcn_wmma_f32_16x16x32_bf16(
              false, a, false, wrel[kc][i], (short)0, acc[i], false, false);
      }
#pragma unroll
      for (int i = 0; i < 4; ++i)
        store_tile_cm_bf16(s_hw + (i * 16) * Nn + m0, lane, acc[i], false);
    }
    __syncthreads();

    // (B) h1 = tanh(adj @ hw + h @ w1_root + b1)
    {
      v8f acc[4];
#pragma unroll
      for (int i = 0; i < 4; ++i)
#pragma unroll
        for (int r = 0; r < 8; ++r) acc[i][r] = bias1[i];

#pragma unroll
      for (int kc = 0; kc < 4; ++kc) {            // adj @ hw, K = 128
        const v16bf a = load_tile_a(s_adj + m0 * Nn + kc * 32, Nn, lane);
#pragma unroll
        for (int i = 0; i < 4; ++i) {
          const v16bf bt = load_tile_b(s_hw + (i * 16) * Nn + kc * 32, Nn, lane);
          acc[i] = __builtin_amdgcn_wmma_f32_16x16x32_bf16(
              false, a, false, bt, (short)0, acc[i], false, false);
        }
      }
#pragma unroll
      for (int kc = 0; kc < 2; ++kc) {            // + h @ w1_root, K = 64
        const v16bf a = load_tile_a(s_hrm + m0 * Hh + kc * 32, Hh, lane);
#pragma unroll
        for (int i = 0; i < 4; ++i)
          acc[i] = __builtin_amdgcn_wmma_f32_16x16x32_bf16(
              false, a, false, wroot[kc][i], (short)0, acc[i], false, false);
      }
#pragma unroll
      for (int i = 0; i < 4; ++i)
        store_tile_cm_bf16(s_h1 + (i * 16) * Nn + m0, lane, acc[i], true);
    }
    __syncthreads();

    // (C) three concurrent jobs (s_hrm has no more readers this step):
    //     tid   0.. 63 : agg2[nn] = adj[nn,:] @ h1  (contiguous in m)
    //     tid  64..127 : NEXT step's row update h[nn+1] = x_{t+1} @ fc_w + fc_b
    //     tid 128..135 : prefetch observation row t+2 (global_prefetch_b8)
    if (tid < Hh) {
      const int j = tid;
      float acc = 0.0f;
      for (int m = 0; m < Nn; ++m)
        acc += (float)s_adj[nn * Nn + m] * (float)s_h1[j * Nn + m];
      s_vec[j] = acc;
    } else if (tid < 2 * Hh) {
      if (t + 1 < Tt) {
        const int j = tid - Hh;
        const int nn_next = (nn0 + t + 1) & (Nn - 1);
        const float* xr = flat + ((size_t)b * Tt + (t + 1)) * Dd;
        float acc = fc_b[j];
        for (int d = 0; d < Dd; ++d) acc += xr[d] * (float)s_fcw_t[j * Dd + d];
        s_hrm[nn_next * Hh + j] = (__bf16)acc;
      }
    } else if (tid < 2 * Hh + 8) {
      if (t + 2 < Tt) {
        const float* pf = flat + ((size_t)b * Tt + (t + 2)) * Dd + (tid - 2 * Hh) * 32;
        __builtin_prefetch(pf, 0, 1);
      }
    }
    __syncthreads();

    // (D) layer-2 row: z2 = agg2_row @ w2_rel + h1[nn] @ w2_root + b2; tanh
    if (tid < Hh) {
      const int j = tid;
      float acc = b2[j];
      for (int k = 0; k < Hh; ++k) {
        acc += s_vec[k] * (float)s_w2rel[k * Hh + j]
             + (float)s_h1[k * Nn + nn] * (float)s_w2root[k * Hh + j];
      }
      s_orow[j] = tanhf(acc);
    }
    __syncthreads();

    // (E) heads: logits (18) + value (1), row b*T + t.
    //     No trailing barrier: next writer of s_orow is (D) of t+1, which is
    //     ordered after this section by the A/B/C barriers of t+1.
    if (tid < NO + 1) {
      float acc;
      if (tid < NO) {
        acc = logit_b[tid];
        for (int k = 0; k < Hh; ++k) acc += s_orow[k] * logit_w[k * NO + tid];
      } else {
        acc = val_b[0];
        for (int k = 0; k < Hh; ++k) acc += s_orow[k] * val_w[k];
      }
      out[((size_t)b * Tt + t) * (NO + 1) + tid] = acc;
    }
  }
}

extern "C" void kernel_launch(void* const* d_in, const int* in_sizes, int n_in,
                              void* d_out, int out_size, void* d_ws, size_t ws_size,
                              hipStream_t stream) {
  const float* flat      = (const float*)d_in[0];
  const float* nodes     = (const float*)d_in[1];
  const float* adj       = (const float*)d_in[2];
  // d_in[3] = weights (unused by reference)
  const int*   num_nodes = (const int*)  d_in[4];
  // d_in[5] = seq_lens (constant T, unused)
  const float* fc_w      = (const float*)d_in[6];
  const float* fc_b      = (const float*)d_in[7];
  const float* w1_rel    = (const float*)d_in[8];
  const float* w1_root   = (const float*)d_in[9];
  const float* b1        = (const float*)d_in[10];
  const float* w2_rel    = (const float*)d_in[11];
  const float* w2_root   = (const float*)d_in[12];
  const float* b2        = (const float*)d_in[13];
  const float* logit_w   = (const float*)d_in[14];
  const float* logit_b   = (const float*)d_in[15];
  const float* val_w     = (const float*)d_in[16];
  const float* val_b     = (const float*)d_in[17];
  float* out = (float*)d_out;

  const int Bsz = in_sizes[4];   // 512 batches -> one workgroup (8 waves) each

  rayobs_graph_scan_kernel<<<dim3(Bsz), dim3(256), 0, stream>>>(
      flat, nodes, adj, num_nodes, fc_w, fc_b,
      w1_rel, w1_root, b1, w2_rel, w2_root, b2,
      logit_w, logit_b, val_w, val_b, out);
}
